// SolarOpen100BAttention_55662776156247
// MI455X (gfx1250) — compile-verified, matrix operand order checked by reading, not run on hardware
//
#include <hip/hip_runtime.h>
#include <hip/hip_bf16.h>

typedef __bf16 bf16;
typedef __attribute__((ext_vector_type(16))) __bf16 v16bf;
typedef __attribute__((ext_vector_type(8)))  float  v8f;
typedef __attribute__((ext_vector_type(4)))  unsigned int u32x4;
typedef __attribute__((ext_vector_type(8)))  unsigned int u32x8;

#define SEQ   2048
#define NH    32
#define NKV   8
#define HD    128
#define HIDN  4096
#define QKDIM (NH * HD)   // 4096
#define KVDIM (NKV * HD)  // 1024

// ---------------- TDM: load a [128 rows x 32 col] bf16 tile (row stride K) into LDS ----------
// D# group0: count=1 | lds_addr | global_addr[56:0] | type=2
// D# group1: data_size=2B, tensor_dim0=K, tensor_dim1=128, tile=32x128, stride0=K
__device__ __forceinline__ void tdm_load_tile(unsigned lds, const bf16* gptr, int K) {
  unsigned long long ga = (unsigned long long)gptr;
  u32x4 g0;
  g0[0] = 1u;                                  // count=1, is_restore=0, gather off
  g0[1] = lds;                                 // lds_addr (bytes)
  g0[2] = (unsigned)ga;                        // global_addr[31:0]
  g0[3] = (unsigned)(ga >> 32) | (2u << 30);   // global_addr[56:32] | type=2
  u32x8 g1;
  g1[0] = 1u << 16;                            // workgroup_mask=0, data_size=1 (2 bytes)
  g1[1] = ((unsigned)K & 0xFFFFu) << 16;       // tensor_dim0[15:0] at bits 63:48
  g1[2] = ((unsigned)K >> 16) | (128u << 16);  // tensor_dim0[31:16] | tensor_dim1[15:0]
  g1[3] = 32u << 16;                           // tensor_dim1[31:16]=0 | tile_dim0=32
  g1[4] = 128u;                                // tile_dim1=128, tile_dim2=0
  g1[5] = (unsigned)K;                         // tensor_dim0_stride[31:0]
  g1[6] = 0u;                                  // stride0[47:32] | stride1[15:0]
  g1[7] = 0u;                                  // stride1[47:16]
  asm volatile("tensor_load_to_lds %0, %1" : : "s"(g0), "s"(g1) : "memory");
}

// ---------------- async global->LDS 16B per lane ----------------
__device__ __forceinline__ void async_ld_b128(unsigned lds, const void* gaddr) {
  asm volatile("global_load_async_to_lds_b128 %0, %1, off" : : "v"(lds), "v"(gaddr) : "memory");
}
__device__ __forceinline__ void wait_asynccnt0() {
  asm volatile("s_wait_asynccnt 0x0" : : : "memory");
}

// ---------------- elementwise converts ----------------
__global__ void f32_to_bf16_k(const float* __restrict__ in, bf16* __restrict__ out, int n) {
  int i = blockIdx.x * blockDim.x + threadIdx.x;
  if (i < n) out[i] = (bf16)in[i];
}

// out is [rows x cols] row-major, in is [cols x rows] row-major: out[r][c] = in[c][r]
__global__ void transpose_to_bf16_k(const float* __restrict__ in, bf16* __restrict__ out,
                                    int rows, int cols) {
  int i = blockIdx.x * blockDim.x + threadIdx.x;
  if (i >= rows * cols) return;
  int r = i / cols, c = i % cols;
  out[i] = (bf16)in[(size_t)c * rows + r];
}

// ---------------- WMMA GEMM: C[M,N] = A[M,K] * B[N,K]^T, TDM double-buffered ---------------
#define BM 128
#define BN 128
#define BK 32

__global__ __launch_bounds__(256) void gemm_bf16_k(const bf16* __restrict__ A,
                                                   const bf16* __restrict__ B,
                                                   float* __restrict__ C,
                                                   int M, int N, int K) {
  __shared__ __attribute__((aligned(32))) bf16 As[2][BM][BK];
  __shared__ __attribute__((aligned(32))) bf16 Bs[2][BN][BK];
  const int tid  = threadIdx.x;
  const int lane = tid & 31;
  const int wid  = tid >> 5;                 // 8 waves
  const int ntn  = N / BN;
  const int m0   = (blockIdx.x / ntn) * BM;
  const int n0   = (blockIdx.x % ntn) * BN;
  const int sm0  = (wid & 3) * 32;           // wave's 32 M-rows
  const int sn0  = (wid >> 2) * 64;          // wave's 64 N-cols
  const int fr   = lane & 15;                // fragment row/col within 16
  const int fk   = (lane >> 4) * 16;         // K half per lane group
  const int ldr  = tid >> 1;                 // prefetch helper row
  const int ldk  = (tid & 1) * 16;

  const bf16* Abase = A + (size_t)m0 * K;
  const bf16* Bbase = B + (size_t)n0 * K;

  v8f acc[2][4] = {};

  if (wid == 0) {  // prologue: tile 0 into buffer 0 via Tensor Data Mover
    tdm_load_tile((unsigned)(size_t)&As[0][0][0], Abase, K);
    tdm_load_tile((unsigned)(size_t)&Bs[0][0][0], Bbase, K);
  }

  int p = 0;
  for (int k0 = 0; k0 < K; k0 += BK) {
    const bool more = (k0 + BK) < K;
    if (wid == 0) {
      if (more) {  // issue next tile pair into the other buffer, wait for current pair
        tdm_load_tile((unsigned)(size_t)&As[p ^ 1][0][0], Abase + (k0 + BK), K);
        tdm_load_tile((unsigned)(size_t)&Bs[p ^ 1][0][0], Bbase + (k0 + BK), K);
        __builtin_amdgcn_s_wait_tensorcnt(2);
      } else {
        __builtin_amdgcn_s_wait_tensorcnt(0);
      }
    }
    if (k0 + 2 * BK < K) {  // warm L2 for the tile after next (global_prefetch_b8)
      __builtin_prefetch(A + (size_t)(m0 + ldr) * K + k0 + 2 * BK + ldk, 0, 0);
      __builtin_prefetch(B + (size_t)(n0 + ldr) * K + k0 + 2 * BK + ldk, 0, 0);
    }
    __syncthreads();

    v16bf af[2], bfr[4];
#pragma unroll
    for (int i = 0; i < 2; ++i) af[i]  = *(const v16bf*)&As[p][sm0 + i * 16 + fr][fk];
#pragma unroll
    for (int j = 0; j < 4; ++j) bfr[j] = *(const v16bf*)&Bs[p][sn0 + j * 16 + fr][fk];
#pragma unroll
    for (int i = 0; i < 2; ++i)
#pragma unroll
      for (int j = 0; j < 4; ++j)
        acc[i][j] = __builtin_amdgcn_wmma_f32_16x16x32_bf16(
            false, af[i], false, bfr[j], (short)0, acc[i][j], false, false);

    __syncthreads();
    p ^= 1;
  }

  const int hf  = lane >> 4;
  const int col = lane & 15;
#pragma unroll
  for (int i = 0; i < 2; ++i)
#pragma unroll
    for (int j = 0; j < 4; ++j) {
      float* cp = C + (size_t)(m0 + sm0 + i * 16 + hf * 8) * N + (n0 + sn0 + j * 16 + col);
#pragma unroll
      for (int v = 0; v < 8; ++v) cp[(size_t)v * N] = acc[i][j][v];
    }
}

// ---------------- YaRN RoPE + repack to [head, S, D] bf16 ----------------
__global__ void rope_pack_k(const float* __restrict__ Qf, const float* __restrict__ Kf,
                            bf16* __restrict__ Qbf, bf16* __restrict__ Kbf) {
  int idx = blockIdx.x * blockDim.x + threadIdx.x;
  const int total = SEQ * (NH + NKV) * (HD / 2);
  if (idx >= total) return;
  const int i    = idx & (HD / 2 - 1);
  const int t    = idx >> 6;
  const int head = t % (NH + NKV);
  const int s    = t / (NH + NKV);

  // YaRN (D=128, base=1e4, factor=4, orig_max=4096):
  //   corr(beta) = D*ln(orig_max/(beta*2*pi)) / (2*ln(base))
  //   low = max(floor(corr(BETA_SLOW=1)),0) = 45 ; high = min(ceil(corr(BETA_FAST=32)),127) = 21
  const float fe = __expf(-((float)(2 * i) / (float)HD) * 9.210340372f); // base^-(2i/D)
  const float fi = fe * 0.25f;                                           // /FACTOR
  float ramp = ((float)i - 45.0f) * (1.0f / (21.0f - 45.0f));
  ramp = fminf(fmaxf(ramp, 0.0f), 1.0f);
  const float invf = fi * ramp + fe * (1.0f - ramp);
  const float ang  = (float)s * invf;
  const float c = __cosf(ang), sn = __sinf(ang);

  if (head < NH) {
    const float* src = Qf + (size_t)s * QKDIM + head * HD;
    bf16* dst = Qbf + ((size_t)head * SEQ + s) * HD;
    float x1 = src[i], x2 = src[i + HD / 2];
    dst[i]          = (bf16)(x1 * c - x2 * sn);
    dst[i + HD / 2] = (bf16)(x2 * c + x1 * sn);
  } else {
    const int kv = head - NH;
    const float* src = Kf + (size_t)s * KVDIM + kv * HD;
    bf16* dst = Kbf + ((size_t)kv * SEQ + s) * HD;
    float x1 = src[i], x2 = src[i + HD / 2];
    dst[i]          = (bf16)(x1 * c - x2 * sn);
    dst[i + HD / 2] = (bf16)(x2 * c + x1 * sn);
  }
}

// ---------------- V: [S, KVDIM] f32 -> [kvh, D, S] bf16 ----------------
__global__ void v_transpose_k(const float* __restrict__ Vf, bf16* __restrict__ Vt) {
  int idx = blockIdx.x * blockDim.x + threadIdx.x;
  if (idx >= SEQ * KVDIM) return;
  int s = idx & (SEQ - 1);
  int c = idx >> 11;  // kvh*HD + d
  Vt[(size_t)c * SEQ + s] = (bf16)Vf[(size_t)s * KVDIM + c];
}

// ---------------- flash attention: one wave per (head, 16 q-rows) ----------------
__global__ __launch_bounds__(256) void flash_attn_k(const bf16* __restrict__ Qbf,
                                                    const bf16* __restrict__ Kbf,
                                                    const bf16* __restrict__ Vt,
                                                    bf16* __restrict__ Obf, float scale) {
  __shared__ __attribute__((aligned(32))) bf16 Pl[8][16][32];   // per-wave P tile
  __shared__ __attribute__((aligned(32))) bf16 Ks[8][32][HD];   // per-wave K tile (async)
  const int lane = threadIdx.x & 31;
  const int wid  = threadIdx.x >> 5;
  const int gw   = blockIdx.x * 8 + wid;
  const int head = gw >> 7;            // SEQ/16 = 128 q-tiles per head
  const int q0   = (gw & 127) * 16;
  const int kvh  = head >> 2;          // GROUPS = 4
  const bf16* Qh = Qbf + ((size_t)head * SEQ + q0) * HD;
  const bf16* Kh = Kbf + (size_t)kvh * SEQ * HD;
  const bf16* Vh = Vt + (size_t)kvh * HD * SEQ;
  const int fr = lane & 15;
  const int hf = lane >> 4;

  const unsigned ldsK = (unsigned)(size_t)&Ks[wid][0][0];

  v16bf qf[4];
#pragma unroll
  for (int kk = 0; kk < 4; ++kk)
    qf[kk] = *(const v16bf*)(Qh + (size_t)fr * HD + kk * 32 + hf * 16);

  v8f o[8] = {};
  float mrow[8], lrow[8];
#pragma unroll
  for (int v = 0; v < 8; ++v) { mrow[v] = -3.0e38f; lrow[v] = 0.0f; }

  const int kend = q0 + 16;  // causal bound (exclusive)
  for (int k0 = 0; k0 < kend; k0 += 32) {
    // --- stage 32x128 bf16 K tile into per-wave LDS via async loads (16B/lane) ---
    const char* gK = (const char*)(Kh + (size_t)k0 * HD);
    const unsigned lanebyte = (unsigned)(hf * 256 + fr * 16);
#pragma unroll
    for (int it = 0; it < 16; ++it)
      async_ld_b128(ldsK + it * 512 + lanebyte, gK + it * 512 + lanebyte);
    wait_asynccnt0();

    float p[2][8];
#pragma unroll
    for (int n = 0; n < 2; ++n) {
      v8f sc = {};
#pragma unroll
      for (int kk = 0; kk < 4; ++kk) {
        v16bf kf = *(const v16bf*)&Ks[wid][n * 16 + fr][kk * 32 + hf * 16];
        sc = __builtin_amdgcn_wmma_f32_16x16x32_bf16(false, qf[kk], false, kf, (short)0, sc,
                                                     false, false);
      }
#pragma unroll
      for (int v = 0; v < 8; ++v) {
        const int qr = q0 + v + 8 * hf;
        const int ky = k0 + n * 16 + fr;
        p[n][v] = (ky <= qr) ? sc[v] * scale : -3.0e38f;
      }
    }
    // online softmax: per-row reductions across 16-lane halves (wave32)
#pragma unroll
    for (int v = 0; v < 8; ++v) {
      float tm = fmaxf(p[0][v], p[1][v]);
#pragma unroll
      for (int off = 1; off < 16; off <<= 1) tm = fmaxf(tm, __shfl_xor(tm, off, 32));
      const float mnew = fmaxf(mrow[v], tm);
      const float corr = __expf(mrow[v] - mnew);
      const float e0 = __expf(p[0][v] - mnew);
      const float e1 = __expf(p[1][v] - mnew);
      float ts = e0 + e1;
#pragma unroll
      for (int off = 1; off < 16; off <<= 1) ts += __shfl_xor(ts, off, 32);
      lrow[v] = lrow[v] * corr + ts;
      mrow[v] = mnew;
      p[0][v] = e0;
      p[1][v] = e1;
#pragma unroll
      for (int dt = 0; dt < 8; ++dt) o[dt][v] = o[dt][v] * corr;
    }
    // redistribute P (C-layout) -> A-operand layout via per-wave LDS bounce
#pragma unroll
    for (int n = 0; n < 2; ++n)
#pragma unroll
      for (int v = 0; v < 8; ++v) Pl[wid][v + 8 * hf][n * 16 + fr] = (bf16)p[n][v];
    v16bf pf = *(const v16bf*)&Pl[wid][fr][hf * 16];
#pragma unroll
    for (int dt = 0; dt < 8; ++dt) {
      v16bf vf = *(const v16bf*)(Vh + (size_t)(dt * 16 + fr) * SEQ + k0 + hf * 16);
      o[dt] = __builtin_amdgcn_wmma_f32_16x16x32_bf16(false, pf, false, vf, (short)0, o[dt],
                                                      false, false);
    }
  }

#pragma unroll
  for (int v = 0; v < 8; ++v) {
    const float inv = 1.0f / lrow[v];
    const int qr = q0 + v + 8 * hf;
    bf16* orow = Obf + (size_t)qr * QKDIM + head * HD;
#pragma unroll
    for (int dt = 0; dt < 8; ++dt) orow[dt * 16 + fr] = (bf16)(o[dt][v] * inv);
  }
}

// ---------------- host launcher ----------------
extern "C" void kernel_launch(void* const* d_in, const int* in_sizes, int n_in,
                              void* d_out, int out_size, void* d_ws, size_t ws_size,
                              hipStream_t stream) {
  (void)in_sizes; (void)n_in; (void)out_size; (void)ws_size;
  const float* x  = (const float*)d_in[0];
  // d_in[1] = position_ids (arange; implied by s index, unused)
  const float* wq = (const float*)d_in[2];
  const float* wk = (const float*)d_in[3];
  const float* wv = (const float*)d_in[4];
  const float* wo = (const float*)d_in[5];
  float* out = (float*)d_out;

  char* ws = (char*)d_ws;
  size_t off = 0;
  auto take = [&](size_t bytes) {
    size_t o = off;
    off += (bytes + 255) & ~(size_t)255;
    return o;
  };
  bf16*  Xbf = (bf16*)(ws + take((size_t)SEQ * HIDN * 2));
  bf16*  Wqb = (bf16*)(ws + take((size_t)QKDIM * HIDN * 2));
  bf16*  Wkb = (bf16*)(ws + take((size_t)KVDIM * HIDN * 2));
  bf16*  Wvb = (bf16*)(ws + take((size_t)KVDIM * HIDN * 2));
  bf16*  Wob = (bf16*)(ws + take((size_t)HIDN * QKDIM * 2));
  float* Qf  = (float*)(ws + take((size_t)SEQ * QKDIM * 4));
  float* Kf  = (float*)(ws + take((size_t)SEQ * KVDIM * 4));
  float* Vf  = (float*)(ws + take((size_t)SEQ * KVDIM * 4));
  bf16*  Qbf = (bf16*)(ws + take((size_t)NH * SEQ * HD * 2));
  bf16*  Kbf = (bf16*)(ws + take((size_t)NKV * SEQ * HD * 2));
  bf16*  Vt  = (bf16*)(ws + take((size_t)NKV * HD * SEQ * 2));
  bf16*  Obf = (bf16*)(ws + take((size_t)SEQ * QKDIM * 2));

  const int TPB = 256;
  auto cdiv = [](int a, int b) { return (a + b - 1) / b; };

  f32_to_bf16_k<<<cdiv(SEQ * HIDN, TPB), TPB, 0, stream>>>(x, Xbf, SEQ * HIDN);
  f32_to_bf16_k<<<cdiv(QKDIM * HIDN, TPB), TPB, 0, stream>>>(wq, Wqb, QKDIM * HIDN);
  f32_to_bf16_k<<<cdiv(KVDIM * HIDN, TPB), TPB, 0, stream>>>(wk, Wkb, KVDIM * HIDN);
  f32_to_bf16_k<<<cdiv(KVDIM * HIDN, TPB), TPB, 0, stream>>>(wv, Wvb, KVDIM * HIDN);
  transpose_to_bf16_k<<<cdiv(HIDN * QKDIM, TPB), TPB, 0, stream>>>(wo, Wob, HIDN, QKDIM);

  gemm_bf16_k<<<(SEQ / BM) * (QKDIM / BN), TPB, 0, stream>>>(Xbf, Wqb, Qf, SEQ, QKDIM, HIDN);
  gemm_bf16_k<<<(SEQ / BM) * (KVDIM / BN), TPB, 0, stream>>>(Xbf, Wkb, Kf, SEQ, KVDIM, HIDN);
  gemm_bf16_k<<<(SEQ / BM) * (KVDIM / BN), TPB, 0, stream>>>(Xbf, Wvb, Vf, SEQ, KVDIM, HIDN);

  rope_pack_k<<<cdiv(SEQ * (NH + NKV) * (HD / 2), TPB), TPB, 0, stream>>>(Qf, Kf, Qbf, Kbf);
  v_transpose_k<<<cdiv(SEQ * KVDIM, TPB), TPB, 0, stream>>>(Vf, Vt);

  flash_attn_k<<<NH * (SEQ / 16) / 8, TPB, 0, stream>>>(Qbf, Kbf, Vt, Obf,
                                                        0.08838834764831845f);

  gemm_bf16_k<<<(SEQ / BM) * (HIDN / BN), TPB, 0, stream>>>(Obf, Wob, out, SEQ, HIDN, QKDIM);
}
